// LstmWorldModel_65524021068071
// MI455X (gfx1250) — matrix operand order repaired and checked by reading, not compiled, not run on hardware
//
#include <hip/hip_runtime.h>

// ---------------- problem constants ----------------
#define T_STEPS 256
#define IN_SIZE 4096
#define ENC_MID 192
#define HIDDEN  2048
#define GATES   8192
#define FULL_W1 9552

// ---------------- WMMA types ----------------
typedef __attribute__((ext_vector_type(16))) __bf16        v16bf;
typedef __attribute__((ext_vector_type(8)))  float         v8f;
typedef __attribute__((ext_vector_type(8)))  unsigned short u16x8;
typedef __attribute__((ext_vector_type(16))) unsigned short u16x16;

struct AFrag { u16x8 lo; u16x8 hi; };   // 32 bytes, matches v16bf

__device__ __forceinline__ unsigned short f2bf(float f) {
  unsigned int u = __builtin_bit_cast(unsigned int, f);
  u += 0x7FFFu + ((u >> 16) & 1u);            // round-to-nearest-even
  return (unsigned short)(u >> 16);
}
__device__ __forceinline__ float bf2f(unsigned short b) {
  return __builtin_bit_cast(float, (unsigned int)b << 16);
}
__device__ __forceinline__ float sigmoidf(float x) { return 1.f / (1.f + __expf(-x)); }

// ---------------- conversion / small kernels ----------------
__global__ void k_cvt_bf16(const float* __restrict__ src, unsigned short* __restrict__ dst, size_t n) {
  size_t i = (size_t)blockIdx.x * blockDim.x + threadIdx.x;
  size_t stride = (size_t)gridDim.x * blockDim.x;
  for (; i < n; i += stride) dst[i] = f2bf(src[i]);
}

// take first dcols of each src row (stride sstride) -> dense bf16 [rows, dcols]
__global__ void k_cvt_rows_bf16(const float* __restrict__ src, unsigned short* __restrict__ dst,
                                int dcols, int sstride, size_t n) {
  size_t i = (size_t)blockIdx.x * blockDim.x + threadIdx.x;
  size_t stride = (size_t)gridDim.x * blockDim.x;
  for (; i < n; i += stride) {
    size_t r = i / dcols, c = i - r * dcols;
    dst[i] = f2bf(src[r * (size_t)sstride + c]);
  }
}

__global__ void k_bias_comb(const float* __restrict__ a, const float* __restrict__ b,
                            float* __restrict__ out, int n) {
  int i = blockIdx.x * blockDim.x + threadIdx.x;
  if (i < n) out[i] = a[i] + b[i];
}

__global__ void k_zero_f32(float* __restrict__ p, size_t n) {
  size_t i = (size_t)blockIdx.x * blockDim.x + threadIdx.x;
  if (i < n) p[i] = 0.f;
}

// ---------------- WMMA fragment loads ----------------
__device__ __forceinline__ v16bf load_afrag(const unsigned short* p) {
  // p already offset by row*K + hi*8; two non-contiguous 16B chunks (K and K+16)
  AFrag af;
  af.lo = *(const u16x8*)(p);
  af.hi = *(const u16x8*)(p + 16);
  return __builtin_bit_cast(v16bf, af);
}
__device__ __forceinline__ v16bf load_bfrag(const unsigned short* p) {
  // p already offset by row*K + hi*16; one contiguous 32B chunk
  return __builtin_bit_cast(v16bf, *(const u16x16*)(p));
}

// ---------------- WMMA GEMM: C[M,N] = act(A[M,K] @ B[N,K]^T + bias) ----------------
// ACT: 0 = identity(+bias), 1 = silu, 2 = tanh(+bias), 3 = *4.0
// Register-blocked: each wave computes a 32x32 tile (2x2 WMMA tiles -> 4 WMMAs
// per 4 fragment loads, 16 flops/byte). 8 waves/block cover 64(M) x 128(N).
template<int ACT, bool OBF>
__global__ void k_gemm_wmma(const unsigned short* __restrict__ A,
                            const unsigned short* __restrict__ B,
                            void* __restrict__ C, const float* __restrict__ bias,
                            int N, int K) {
  const int wave = threadIdx.x >> 5;        // 0..7
  const int lane = threadIdx.x & 31;
  const int wn   = wave & 3;                // N subtile
  const int wm   = wave >> 2;               // M subtile
  const int tn0  = (blockIdx.x * 4 + wn) * 32;
  if (tn0 >= N) return;                     // wave-uniform: EXEC stays all-1s
  const int tm0  = (blockIdx.y * 2 + wm) * 32;
  const int r    = lane & 15;
  const int hi   = lane >> 4;

  // A fragment rows: one contiguous 16-elem chunk at K offset hi*16
  const unsigned short* a0 = A + (size_t)(tm0 + r) * K + hi * 16;
  const unsigned short* a1 = a0 + (size_t)16 * K;
  // B fragment rows: one contiguous 16-elem chunk at K offset hi*16
  const unsigned short* b0 = B + (size_t)(tn0 + r) * K + hi * 16;
  const unsigned short* b1 = b0 + (size_t)16 * K;

  v8f acc00 = {}, acc01 = {}, acc10 = {}, acc11 = {};
  for (int k0 = 0; k0 < K; k0 += 32) {
    const v16bf fa0 = load_bfrag(a0 + k0);
    const v16bf fa1 = load_bfrag(a1 + k0);
    const v16bf fb0 = load_bfrag(b0 + k0);
    const v16bf fb1 = load_bfrag(b1 + k0);
    acc00 = __builtin_amdgcn_wmma_f32_16x16x32_bf16(false, fa0, false, fb0, (short)0, acc00, false, false);
    acc01 = __builtin_amdgcn_wmma_f32_16x16x32_bf16(false, fa0, false, fb1, (short)0, acc01, false, false);
    acc10 = __builtin_amdgcn_wmma_f32_16x16x32_bf16(false, fa1, false, fb0, (short)0, acc10, false, false);
    acc11 = __builtin_amdgcn_wmma_f32_16x16x32_bf16(false, fa1, false, fb1, (short)0, acc11, false, false);
  }

  // C layout per tile: lane -> column n = tn+(lane&15); VGPR r2 -> row tm + r2 + 8*hi
  const float bv0 = bias ? bias[tn0 + r]      : 0.f;
  const float bv1 = bias ? bias[tn0 + 16 + r] : 0.f;

  auto store_tile = [&](const v8f& acc, int tm, int tn, float bval) {
    const int n = tn + r;
#pragma unroll
    for (int r2 = 0; r2 < 8; ++r2) {
      const int m = tm + r2 + hi * 8;
      float v = acc[r2] + bval;
      if (ACT == 1)      v = v * sigmoidf(v);
      else if (ACT == 2) v = tanhf(v);
      else if (ACT == 3) v = v * 4.0f;
      const size_t idx = (size_t)m * N + n;
      if (OBF) ((unsigned short*)C)[idx] = f2bf(v);
      else     ((float*)C)[idx] = v;
    }
  };
  store_tile(acc00, tm0,      tn0,      bv0);
  store_tile(acc01, tm0,      tn0 + 16, bv1);
  store_tile(acc10, tm0 + 16, tn0,      bv0);
  store_tile(acc11, tm0 + 16, tn0 + 16, bv1);
}

// ---------------- LSTM recurrence (sequential) ----------------
// gates[row] = preG[row] + dot(w_hh_bf16[row,:], h_f32) — one wave per row of 8192.
// w_hh in bf16 halves the dominant per-step L2 stream (67MB -> 33.5MB/step).
__global__ void k_lstm_matvec(const unsigned short* __restrict__ whh_bf, const float* __restrict__ h,
                              const float* __restrict__ preG, float* __restrict__ gates) {
  const int gwave = (int)((blockIdx.x * blockDim.x + threadIdx.x) >> 5);
  const int lane  = threadIdx.x & 31;
  if (gwave >= GATES) return;
  const u16x8*  wr = (const u16x8*)(whh_bf + (size_t)gwave * HIDDEN);
  const float4* hv = (const float4*)h;
  float s = 0.f;
#pragma unroll 2
  for (int i = lane; i < HIDDEN / 8; i += 32) {   // 8 iters: 16B of weights/lane/iter
    const u16x8 w = wr[i];
    const float4 x0 = hv[2 * i];
    const float4 x1 = hv[2 * i + 1];
    s += bf2f(w[0]) * x0.x + bf2f(w[1]) * x0.y + bf2f(w[2]) * x0.z + bf2f(w[3]) * x0.w;
    s += bf2f(w[4]) * x1.x + bf2f(w[5]) * x1.y + bf2f(w[6]) * x1.z + bf2f(w[7]) * x1.w;
  }
#pragma unroll
  for (int off = 16; off > 0; off >>= 1) s += __shfl_xor(s, off, 32);
  if (lane == 0) gates[gwave] = preG[gwave] + s;
}

__global__ void k_lstm_update(const float* __restrict__ gates, float* __restrict__ h,
                              float* __restrict__ c, float* __restrict__ sout) {
  const int j = blockIdx.x * blockDim.x + threadIdx.x;
  if (j >= HIDDEN) return;
  const float ig = sigmoidf(gates[j]);
  const float fg = sigmoidf(gates[HIDDEN + j]);
  const float g  = tanhf(gates[2 * HIDDEN + j]);
  const float og = sigmoidf(gates[3 * HIDDEN + j]);
  const float cn = fg * c[j] + ig * g;
  const float hn = og * tanhf(cn);
  c[j] = cn;
  h[j] = hn;
  sout[j] = hn * 4.0f;   // OUT_TEMP
}

// ---------------- per-variable softmax over 32 states -> bf16 probs ----------------
__global__ void k_softmax32(const float* __restrict__ s, unsigned short* __restrict__ p, int nwaves) {
  const int gwave = (int)((blockIdx.x * blockDim.x + threadIdx.x) >> 5);
  const int lane  = threadIdx.x & 31;
  if (gwave >= nwaves) return;
  float x = s[(size_t)gwave * 32 + lane];
  float m = x;
#pragma unroll
  for (int off = 16; off > 0; off >>= 1) m = fmaxf(m, __shfl_xor(m, off, 32));
  float e = __expf(x - m);
  float sum = e;
#pragma unroll
  for (int off = 16; off > 0; off >>= 1) sum += __shfl_xor(sum, off, 32);
  p[(size_t)gwave * 32 + lane] = f2bf(e / sum);
}

// ---------------- host-side helpers ----------------
template<int ACT, bool OBF>
static void run_gemm(const unsigned short* A, const unsigned short* B, void* C,
                     const float* bias, int M, int N, int K, hipStream_t stream) {
  dim3 grid((N + 127) / 128, M / 64);   // block = 8 waves = 64(M) x 128(N) tile
  k_gemm_wmma<ACT, OBF><<<grid, 256, 0, stream>>>(A, B, C, bias, N, K);
}

static void run_cvt(const float* src, unsigned short* dst, size_t n, hipStream_t stream) {
  int blocks = (int)((n + 255) / 256);
  if (blocks > 4096) blocks = 4096;
  k_cvt_bf16<<<blocks, 256, 0, stream>>>(src, dst, n);
}

extern "C" void kernel_launch(void* const* d_in, const int* in_sizes, int n_in,
                              void* d_out, int out_size, void* d_ws, size_t ws_size,
                              hipStream_t stream) {
  (void)in_sizes; (void)n_in; (void)out_size; (void)ws_size;

  const float* obs    = (const float*)d_in[0];   // [256, 4096]
  const float* enc_w1 = (const float*)d_in[1];   // [192, 9552]
  const float* enc_w2 = (const float*)d_in[2];   // [2048, 192]
  const float* enc_b2 = (const float*)d_in[3];   // [2048]
  const float* w_ih   = (const float*)d_in[4];   // [8192, 2048]
  const float* w_hh   = (const float*)d_in[5];   // [8192, 2048]
  const float* b_ih   = (const float*)d_in[6];   // [8192]
  const float* b_hh   = (const float*)d_in[7];   // [8192]
  const float* dec_w1 = (const float*)d_in[8];   // [2048, 2048]
  const float* dec_w2 = (const float*)d_in[9];   // [4096, 2048]

  // ---- bump allocator over d_ws (256B aligned) ----
  char* base = (char*)d_ws;
  size_t off = 0;
  auto alloc = [&](size_t bytes) -> void* {
    off = (off + 255) & ~(size_t)255;
    void* p = base + off;
    off += bytes;
    return p;
  };

  unsigned short* obs_bf = (unsigned short*)alloc((size_t)T_STEPS * IN_SIZE * 2);
  unsigned short* w1_bf  = (unsigned short*)alloc((size_t)ENC_MID * IN_SIZE * 2);
  unsigned short* w2_bf  = (unsigned short*)alloc((size_t)HIDDEN * ENC_MID * 2);
  unsigned short* wih_bf = (unsigned short*)alloc((size_t)GATES * HIDDEN * 2);
  unsigned short* whh_bf = (unsigned short*)alloc((size_t)GATES * HIDDEN * 2);
  unsigned short* dw1_bf = (unsigned short*)alloc((size_t)HIDDEN * HIDDEN * 2);
  unsigned short* dw2_bf = (unsigned short*)alloc((size_t)IN_SIZE * HIDDEN * 2);
  unsigned short* mid_bf = (unsigned short*)alloc((size_t)T_STEPS * ENC_MID * 2);
  unsigned short* enc_bf = (unsigned short*)alloc((size_t)T_STEPS * HIDDEN * 2);
  float* preGates        = (float*)alloc((size_t)T_STEPS * GATES * 4);
  float* bcomb           = (float*)alloc((size_t)GATES * 4);
  float* hc              = (float*)alloc((size_t)2 * HIDDEN * 4);
  float* gates           = (float*)alloc((size_t)GATES * 4);
  float* souts           = (float*)alloc((size_t)T_STEPS * HIDDEN * 4);
  unsigned short* prob_bf = (unsigned short*)alloc((size_t)T_STEPS * HIDDEN * 2);
  unsigned short* hid_bf  = (unsigned short*)alloc((size_t)T_STEPS * HIDDEN * 2);
  float* h = hc;
  float* c = hc + HIDDEN;

  // ---- stage 0: weight / input conversion to bf16 ----
  run_cvt(obs, obs_bf, (size_t)T_STEPS * IN_SIZE, stream);
  {
    size_t n = (size_t)ENC_MID * IN_SIZE;
    int blocks = (int)((n + 255) / 256); if (blocks > 4096) blocks = 4096;
    k_cvt_rows_bf16<<<blocks, 256, 0, stream>>>(enc_w1, w1_bf, IN_SIZE, FULL_W1, n);
  }
  run_cvt(enc_w2, w2_bf, (size_t)HIDDEN * ENC_MID, stream);
  run_cvt(w_ih,   wih_bf, (size_t)GATES * HIDDEN, stream);
  run_cvt(w_hh,   whh_bf, (size_t)GATES * HIDDEN, stream);
  run_cvt(dec_w1, dw1_bf, (size_t)HIDDEN * HIDDEN, stream);
  run_cvt(dec_w2, dw2_bf, (size_t)IN_SIZE * HIDDEN, stream);
  k_bias_comb<<<GATES / 256, 256, 0, stream>>>(b_ih, b_hh, bcomb, GATES);
  k_zero_f32<<<(2 * HIDDEN + 255) / 256, 256, 0, stream>>>(hc, 2 * HIDDEN);

  // ---- stage 1: encoder ----
  // mid = silu(obs @ enc_w1[:, :4096]^T)        [256, 192]
  run_gemm<1, true>(obs_bf, w1_bf, mid_bf, nullptr, T_STEPS, ENC_MID, IN_SIZE, stream);
  // enc = tanh(mid @ enc_w2^T + enc_b2)         [256, 2048]
  run_gemm<2, true>(mid_bf, w2_bf, enc_bf, enc_b2, T_STEPS, HIDDEN, ENC_MID, stream);

  // ---- stage 2: hoisted input projection ----
  // preGates = enc @ w_ih^T + (b_ih + b_hh)     [256, 8192]  (f32)
  run_gemm<0, false>(enc_bf, wih_bf, preGates, bcomb, T_STEPS, GATES, HIDDEN, stream);

  // ---- stage 3: sequential LSTM recurrence ----
  for (int t = 0; t < T_STEPS; ++t) {
    k_lstm_matvec<<<GATES * 32 / 256, 256, 0, stream>>>(whh_bf, h, preGates + (size_t)t * GATES, gates);
    k_lstm_update<<<(HIDDEN + 255) / 256, 256, 0, stream>>>(gates, h, c, souts + (size_t)t * HIDDEN);
  }

  // ---- stage 4: decoder ----
  // probs = softmax over groups of 32 of (h * 4)
  {
    int nwaves = T_STEPS * (HIDDEN / 32);
    k_softmax32<<<nwaves * 32 / 256, 256, 0, stream>>>(souts, prob_bf, nwaves);
  }
  // hid = silu(probs @ dec_w1^T)                [256, 2048]
  run_gemm<1, true>(prob_bf, dw1_bf, hid_bf, nullptr, T_STEPS, HIDDEN, HIDDEN, stream);
  // out = (hid @ dec_w2^T) * 4                  [256, 4096]  (f32, straight to d_out)
  run_gemm<3, false>(hid_bf, dw2_bf, d_out, nullptr, T_STEPS, IN_SIZE, HIDDEN, stream);
}
